// ModelPaperBaseline_bin_33036888441366
// MI455X (gfx1250) — matrix operand order repaired and checked by reading, not compiled, not run on hardware
//
#include <hip/hip_runtime.h>
#include <hip/hip_bf16.h>
#include <math.h>

// ---------------------------------------------------------------------------
// CDNA5 / gfx1250 quantized CNN+MLP forward.
// wave32, v_wmma_f32_16x16x32_f16 for all conv/FC GEMMs, f32 accumulate.
// Train-mode BN => per-layer global stats; the normalize+actq+residual of
// layer i is FUSED into the conv of layer i+1 (h is never materialized for
// the residual chain): h_i = actq(bn_i(y_i)) + actq(bn_0(y_0)).  Each wave
// normalizes its own row once, stages the 16x32 h tile in a per-wave LDS
// tile (ds_store_b128), and builds the 3 shifted A fragments via
// ds_load_b128 (wave-local ordering via s_wait_dscnt 0).
// Per residual block DRAM traffic: 96MB (vs 160MB unfused).
// Weights: 1-bit (sign*mean|w|), pre-swizzled into exact WMMA B-fragment
// lane layout so each lane loads its fragment as one contiguous 32B.
// ---------------------------------------------------------------------------

typedef __attribute__((ext_vector_type(16))) _Float16 v16h;
typedef __attribute__((ext_vector_type(8)))  _Float16 v8h;
typedef __attribute__((ext_vector_type(8)))  float    v8f;

#define BATCH   32768
#define EPSBN   0.01f
#define NQF     255.0f

union V16u { v16h v; v8h h[2]; };

__device__ __forceinline__ v16h hzero16() {
  v16h z;
#pragma unroll
  for (int i = 0; i < 16; ++i) z[i] = (_Float16)0.0f;
  return z;
}

__device__ __forceinline__ float normq1(float y, float sc, float bi) {
  float v = y * sc + bi;
  v = fminf(fmaxf(v, 0.f), 1.f);          // relu + clip(0,1)
  return rintf(v * NQF) * (1.f / NQF);    // RNE round, matches jnp.round
}

// ---------------------------------------------------------------------------
// Weight prep: per-tensor E = mean|w|, quantize to sign(w)*E, swizzle into
// WMMA B-fragment layout (frag elem e of lane L = row ((L>=16)?8:0)+(e<8?e:e+8),
// col L&15; lane L reads 16 f16 at offset L*16).
// Blocks: [0]=conv0, [1..9]=convs, [10]=fc1(permuted), [11]=fc2, [12]=fc3.
// ---------------------------------------------------------------------------
__global__ void prep_weights_kernel(const float* __restrict__ conv0_w,
                                    const float* __restrict__ convs_w,
                                    const float* __restrict__ fc1_w,
                                    const float* __restrict__ fc2_w,
                                    const float* __restrict__ fc3_w,
                                    float* __restrict__ qw0,
                                    _Float16* __restrict__ wqc,
                                    _Float16* __restrict__ wq1,
                                    _Float16* __restrict__ wq2,
                                    float* __restrict__ qw3) {
  __shared__ float red[256];
  const int tid = threadIdx.x;
  const int bid = blockIdx.x;
  const float* src;
  int n;
  if (bid == 0)       { src = conv0_w;                 n = 128;   }
  else if (bid <= 9)  { src = convs_w + (bid-1)*3072;  n = 3072;  }
  else if (bid == 10) { src = fc1_w;                   n = 32768; }
  else if (bid == 11) { src = fc2_w;                   n = 4096;  }
  else                { src = fc3_w;                   n = 64;    }

  float s = 0.f;
  for (int i = tid; i < n; i += 256) s += fabsf(src[i]);
  red[tid] = s;
  __syncthreads();
  for (int w = 128; w > 0; w >>= 1) {
    if (tid < w) red[tid] += red[tid + w];
    __syncthreads();
  }
  const float E = red[0] / (float)n;

  if (bid == 0) {
    for (int i = tid; i < 128; i += 256) {
      float w = src[i];
      qw0[i] = (w > 0.f) ? E : ((w < 0.f) ? -E : 0.f);
    }
  } else if (bid <= 9) {
    const int blk = bid - 1;
    for (int idx = tid; idx < 3072; idx += 256) {
      int f = idx >> 9, r = idx & 511;
      int tap = f >> 1, t = f & 1;
      int L = r >> 4, e = r & 15;
      int row = ((L >> 4) << 3) + (e < 8 ? e : e + 8);  // input channel 0..31
      int o   = t * 16 + (L & 15);            // output channel
      float w = convs_w[((blk*32 + o)*32 + row)*3 + tap];
      wqc[blk*3072 + idx] = (_Float16)((w > 0.f) ? E : ((w < 0.f) ? -E : 0.f));
    }
  } else if (bid == 10) {
    // fc1 weights permuted: activations are [b][pos][ch]; reference flattens
    // channel-major (K = c*16 + l)  =>  Kp = l*32 + c, l = kstep.
    for (int idx = tid; idx < 32768; idx += 256) {
      int f = idx >> 9, r = idx & 511;
      int nt = f >> 4, k = f & 15;
      int L = r >> 4, e = r & 15;
      int row = ((L >> 4) << 3) + (e < 8 ? e : e + 8);  // channel c
      int o   = nt * 16 + (L & 15);
      float w = fc1_w[o*512 + row*16 + k];
      wq1[idx] = (_Float16)((w > 0.f) ? E : ((w < 0.f) ? -E : 0.f));
    }
  } else if (bid == 11) {
    for (int idx = tid; idx < 4096; idx += 256) {
      int f = idx >> 9, r = idx & 511;
      int nt = f >> 1, k = f & 1;
      int L = r >> 4, e = r & 15;
      int row = ((L >> 4) << 3) + (e < 8 ? e : e + 8);
      int o   = nt * 16 + (L & 15);
      float w = fc2_w[o*64 + k*32 + row];
      wq2[idx] = (_Float16)((w > 0.f) ? E : ((w < 0.f) ? -E : 0.f));
    }
  } else {
    for (int i = tid; i < 64; i += 256) {
      float w = src[i];
      qw3[i] = (w > 0.f) ? E : ((w < 0.f) ? -E : 0.f);
    }
  }
}

// ---------------------------------------------------------------------------
// conv0 (1x1, Cin=4): scalar VALU (trivial FLOPs). lane = output channel.
// Writes pre-BN y0 to Y0[b][l][ch] f16 + per-channel stats. Bias cancels
// under train-mode BN.
// ---------------------------------------------------------------------------
__global__ void conv0_kernel(const float* __restrict__ x,
                             const float* __restrict__ qw0,
                             _Float16* __restrict__ Y,
                             float* __restrict__ stats) {
  const int lane = threadIdx.x & 31;
  const int wave = (blockIdx.x * blockDim.x + threadIdx.x) >> 5;
  const int nw   = (gridDim.x * blockDim.x) >> 5;
  const float w0 = qw0[lane*4+0], w1 = qw0[lane*4+1];
  const float w2 = qw0[lane*4+2], w3 = qw0[lane*4+3];
  float sum = 0.f, ss = 0.f;
  for (int p = wave; p < BATCH*16; p += nw) {
    const int b = p >> 4, l = p & 15;
    const float* xb = x + b*64 + l;
    float y = w0*xb[0] + w1*xb[16] + w2*xb[32] + w3*xb[48];
    Y[b*512 + l*32 + lane] = (_Float16)y;
    sum += y; ss += y*y;
  }
  atomicAdd(&stats[lane], sum);
  atomicAdd(&stats[32 + lane], ss);
}

// ---------------------------------------------------------------------------
// Fused residual conv block:
//   h = normq(Yp, sbp) [+ normq(Y0, sb0) if add0]   (built inline, staged in
//   a per-wave 1KB LDS tile), then y_out = conv3(h) via WMMA:
//   M=16 positions, K=96 (3 taps x 32 ch), N=32 ch (2 tiles), 6 WMMAs/sample.
// In-place Yout==Yp is safe: each sample is owned by one wave and all its
// loads are issued before its stores.
// Channel stats accumulated in registers; 4 atomics/lane at kernel end.
// ---------------------------------------------------------------------------
__global__ void conv_wmma_fused_kernel(const _Float16* __restrict__ Yp,
                                       const _Float16* __restrict__ Y0,
                                       const float* __restrict__ sbp,
                                       const float* __restrict__ sb0,
                                       const _Float16* __restrict__ WQ, // 6*512
                                       _Float16* __restrict__ Yout,
                                       float* __restrict__ stats,
                                       int add0) {
  extern __shared__ _Float16 ldsbuf[];
  const int lane  = threadIdx.x & 31;
  _Float16* tile  = ldsbuf + (threadIdx.x >> 5) * 512;   // per-wave 16x32 tile
  const int wave  = (blockIdx.x * blockDim.x + threadIdx.x) >> 5;
  const int nw    = (gridDim.x * blockDim.x) >> 5;
  const int row   = lane & 15;
  const int c0    = (lane < 16) ? 0 : 8;
  const int r0    = c0;

  // Per-lane scale/bias for its 16 elements (elem j -> ch c0 + (j<8?j:j+8))
  float scp[16], bip[16], sc0[16], bi0[16];
#pragma unroll
  for (int j = 0; j < 16; ++j) {
    const int ch = c0 + ((j < 8) ? j : j + 8);
    scp[j] = sbp[ch]; bip[j] = sbp[32 + ch];
    sc0[j] = sb0[ch]; bi0[j] = sb0[32 + ch];
  }

  v16h bf[6];
#pragma unroll
  for (int f = 0; f < 6; ++f) bf[f] = *(const v16h*)(WQ + f*512 + lane*16);

  const v16h zfrag = hzero16();
  float sum0 = 0.f, ss0 = 0.f, sum1 = 0.f, ss1 = 0.f;

  for (int b = wave; b < BATCH; b += nw) {
    const _Float16* ypb = Yp + (size_t)b * 512 + row * 32;
    const _Float16* y0b = Y0 + (size_t)b * 512 + row * 32;
    if (b + nw < BATCH)
      __builtin_prefetch(Yp + (size_t)(b + nw) * 512 + row * 32, 0, 1);

    // normalize this lane's 16 h values (each h value computed exactly once
    // per wave: lanes L and L+16 cover complementary channel chunks)
    const v8h ya = *(const v8h*)(ypb + c0);
    const v8h yb = *(const v8h*)(ypb + c0 + 16);
    v8h za, zb;
    if (add0) { za = *(const v8h*)(y0b + c0); zb = *(const v8h*)(y0b + c0 + 16); }
    v8h h0v, h1v;
#pragma unroll
    for (int j = 0; j < 16; ++j) {
      const float y = (j < 8) ? (float)ya[j] : (float)yb[j - 8];
      float v = normq1(y, scp[j], bip[j]);
      if (add0) {
        const float y0 = (j < 8) ? (float)za[j] : (float)zb[j - 8];
        v += normq1(y0, sc0[j], bi0[j]);
      }
      if (j < 8) h0v[j] = (_Float16)v; else h1v[j - 8] = (_Float16)v;
    }
    // stage tile in LDS (wave-local)
    *(v8h*)(tile + row*32 + c0)      = h0v;
    *(v8h*)(tile + row*32 + c0 + 16) = h1v;
    asm volatile("s_wait_dscnt 0" ::: "memory");

    // build the 3 shifted A fragments from LDS
    v16h af[3];
#pragma unroll
    for (int s = 0; s < 3; ++s) {
      const int l = row + s - 1;
      if (l >= 0 && l < 16) {
        V16u u;
        u.h[0] = *(const v8h*)(tile + l*32 + c0);
        u.h[1] = *(const v8h*)(tile + l*32 + c0 + 16);
        af[s] = u.v;
      } else {
        af[s] = zfrag;
      }
    }

    _Float16* yo = Yout + (size_t)b * 512;
#pragma unroll
    for (int t = 0; t < 2; ++t) {
      v8f acc = {};
      acc = __builtin_amdgcn_wmma_f32_16x16x32_f16(false, af[0], false, bf[0 + t],
                                                   (short)0, acc, false, false);
      acc = __builtin_amdgcn_wmma_f32_16x16x32_f16(false, af[1], false, bf[2 + t],
                                                   (short)0, acc, false, false);
      acc = __builtin_amdgcn_wmma_f32_16x16x32_f16(false, af[2], false, bf[4 + t],
                                                   (short)0, acc, false, false);
      const int ch = t*16 + row;
#pragma unroll
      for (int r = 0; r < 8; ++r) {
        const float v = acc[r];
        if (t == 0) { sum0 += v; ss0 += v*v; } else { sum1 += v; ss1 += v*v; }
        yo[(r0 + r)*32 + ch] = (_Float16)v;
      }
    }
  }
  atomicAdd(&stats[row],      sum0);
  atomicAdd(&stats[32 + row], ss0);
  atomicAdd(&stats[16 + row], sum1);
  atomicAdd(&stats[48 + row], ss1);
}

// ---------------------------------------------------------------------------
// FC GEMM (N=64 -> 4 tiles): fc1 (K=512) / fc2 (K=64). A rows contiguous.
// ---------------------------------------------------------------------------
__global__ void fc_wmma_kernel(const _Float16* __restrict__ A,
                               const _Float16* __restrict__ WQ,
                               _Float16* __restrict__ Y,
                               float* __restrict__ stats,
                               int ksteps, int kdim) {
  const int lane = threadIdx.x & 31;
  const int wave = (blockIdx.x * blockDim.x + threadIdx.x) >> 5;
  const int nw   = (gridDim.x * blockDim.x) >> 5;
  const int row  = lane & 15;
  const int c0   = (lane < 16) ? 0 : 8;
  const int r0   = c0;
  const int mtiles = BATCH / 16;

  for (int mt = wave; mt < mtiles; mt += nw) {
    const int b0 = mt * 16;
    const _Float16* ab = A + (size_t)(b0 + row) * kdim;
#pragma unroll
    for (int nt = 0; nt < 4; ++nt) {
      v8f acc = {};
      for (int k = 0; k < ksteps; ++k) {
        V16u u;
        u.h[0] = *(const v8h*)(ab + k*32 + c0);
        u.h[1] = *(const v8h*)(ab + k*32 + c0 + 16);
        const v16h bfrag = *(const v16h*)(WQ + (size_t)(nt*ksteps + k)*512 + lane*16);
        acc = __builtin_amdgcn_wmma_f32_16x16x32_f16(false, u.v, false, bfrag,
                                                     (short)0, acc, false, false);
      }
      const int o = nt*16 + row;
      float sum = 0.f, ss = 0.f;
#pragma unroll
      for (int r = 0; r < 8; ++r) {
        const float v = acc[r];
        sum += v; ss += v*v;
        Y[(size_t)(b0 + r0 + r)*64 + o] = (_Float16)v;
      }
      atomicAdd(&stats[o], sum);
      atomicAdd(&stats[64 + o], ss);
    }
  }
}

// ---------------------------------------------------------------------------
// BN finalize: (sum, sumsq) -> (scale, bias).
// ---------------------------------------------------------------------------
__global__ void bn_finalize_kernel(const float* __restrict__ stats,
                                   const float* __restrict__ g,
                                   const float* __restrict__ b,
                                   float* __restrict__ sb,
                                   int nch, float invN) {
  const int c = threadIdx.x;
  if (c < nch) {
    const float m  = stats[c] * invN;
    const float va = stats[nch + c] * invN - m*m;
    const float sc = g[c] * rsqrtf(va + EPSBN);
    sb[c]       = sc;
    sb[nch + c] = b[c] - m*sc;
  }
}

// ---------------------------------------------------------------------------
// Single-stream normalize+actq (fc outputs).
// ---------------------------------------------------------------------------
__global__ void norm_actq_kernel(const _Float16* __restrict__ Y,
                                 _Float16* __restrict__ Out,
                                 const float* __restrict__ sb,
                                 int nch, int nchunks) {
  const int t = blockIdx.x * blockDim.x + threadIdx.x;
  if (t >= nchunks) return;
  const size_t g = (size_t)t * 8;
  const int ch0 = (int)(g & (size_t)(nch - 1));
  v8h y = *(const v8h*)(Y + g);
  v8h o;
#pragma unroll
  for (int j = 0; j < 8; ++j)
    o[j] = (_Float16)normq1((float)y[j], sb[ch0 + j], sb[nch + ch0 + j]);
  *(v8h*)(Out + g) = o;
}

// ---------------------------------------------------------------------------
// Two-stream normalize+actq+add: materialize final h for fc1:
//   H = normq(Yw, sbp) + normq(Y0, sb0)
// ---------------------------------------------------------------------------
__global__ void norm2_kernel(const _Float16* __restrict__ Yp,
                             const _Float16* __restrict__ Y0,
                             _Float16* __restrict__ Out,
                             const float* __restrict__ sbp,
                             const float* __restrict__ sb0,
                             int nchunks) {
  const int t = blockIdx.x * blockDim.x + threadIdx.x;
  if (t >= nchunks) return;
  const size_t g = (size_t)t * 8;
  const int ch0 = (int)(g & 31u);
  v8h yp = *(const v8h*)(Yp + g);
  v8h y0 = *(const v8h*)(Y0 + g);
  v8h o;
#pragma unroll
  for (int j = 0; j < 8; ++j) {
    float v = normq1((float)yp[j], sbp[ch0 + j], sbp[32 + ch0 + j])
            + normq1((float)y0[j], sb0[ch0 + j], sb0[32 + ch0 + j]);
    o[j] = (_Float16)v;
  }
  *(v8h*)(Out + g) = o;
}

// ---------------------------------------------------------------------------
// fc3 + sigmoid.
// ---------------------------------------------------------------------------
__global__ void fc3_sigmoid_kernel(const _Float16* __restrict__ A2,
                                   const float* __restrict__ qw3,
                                   const float* __restrict__ fc3_b,
                                   float* __restrict__ out) {
  const int b = blockIdx.x * blockDim.x + threadIdx.x;
  if (b >= BATCH) return;
  const _Float16* a = A2 + (size_t)b * 64;
  float z = fc3_b[0];
#pragma unroll 8
  for (int j = 0; j < 64; ++j) z += (float)a[j] * qw3[j];
  out[b] = 1.f / (1.f + expf(-z));
}

// ---------------------------------------------------------------------------
extern "C" void kernel_launch(void* const* d_in, const int* in_sizes, int n_in,
                              void* d_out, int out_size, void* d_ws, size_t ws_size,
                              hipStream_t stream) {
  const float* x       = (const float*)d_in[0];
  const float* conv0_w = (const float*)d_in[1];
  const float* bn0_g   = (const float*)d_in[3];
  const float* bn0_b   = (const float*)d_in[4];
  const float* convs_w = (const float*)d_in[5];
  const float* bns_g   = (const float*)d_in[7];
  const float* bns_b   = (const float*)d_in[8];
  const float* fc1_w   = (const float*)d_in[9];
  const float* bn5_g   = (const float*)d_in[11];
  const float* bn5_b   = (const float*)d_in[12];
  const float* fc2_w   = (const float*)d_in[13];
  const float* bn6_g   = (const float*)d_in[15];
  const float* bn6_b   = (const float*)d_in[16];
  const float* fc3_w   = (const float*)d_in[17];
  const float* fc3_b   = (const float*)d_in[18];
  float* out = (float*)d_out;

  // Workspace (~112MB big buffers + ~0.2MB small)
  char* ws = (char*)d_ws;
  const size_t MB = 1024ull * 1024ull;
  _Float16* Y0 = (_Float16*)(ws);              // conv0 pre-BN, [B][16][32], 32MB
  _Float16* Yw = (_Float16*)(ws + 32*MB);      // working pre-BN (in-place), 32MB
  _Float16* H  = (_Float16*)(ws + 64*MB);      // final h for fc1, 32MB
  _Float16* Y1 = (_Float16*)(ws + 96*MB);      // fc1 pre-BN, [B][64], 4MB
  _Float16* A1 = (_Float16*)(ws + 100*MB);
  _Float16* Y2 = (_Float16*)(ws + 104*MB);
  _Float16* A2 = (_Float16*)(ws + 108*MB);
  char* sm = ws + 112*MB;
  float*    qw0   = (float*)sm;                sm += 512;
  _Float16* wqc   = (_Float16*)sm;             sm += 9*3072*2;
  _Float16* wq1   = (_Float16*)sm;             sm += 32768*2;
  _Float16* wq2   = (_Float16*)sm;             sm += 4096*2;
  float*    qw3   = (float*)sm;                sm += 256;
  float*    stats = (float*)sm;                sm += 12*128*4;  // [layer][sum|ss]
  float*    sb    = (float*)sm;                sm += 12*128*4;  // [layer][scale|bias]

  const float invNconv = 1.f / (float)(BATCH * 16);
  const float invNfc   = 1.f / (float)BATCH;

  hipMemsetAsync(stats, 0, 12*128*sizeof(float), stream);

  prep_weights_kernel<<<13, 256, 0, stream>>>(conv0_w, convs_w, fc1_w, fc2_w, fc3_w,
                                              qw0, wqc, wq1, wq2, qw3);

  // ---- conv0 -> stats -> sb0 (h0 never materialized) ----
  conv0_kernel<<<512, 256, 0, stream>>>(x, qw0, Y0, stats);
  bn_finalize_kernel<<<1, 64, 0, stream>>>(stats, bn0_g, bn0_b, sb, 32, invNconv);

  // ---- 9 residual blocks, norm-of-prev fused into conv ----
  for (int i = 0; i < 9; ++i) {
    const _Float16* Yp  = (i == 0) ? Y0 : Yw;
    const float*    sbp = (i == 0) ? sb : (sb + i*128);   // prev layer's scale/bias
    float* st = stats + (1 + i)*128;
    conv_wmma_fused_kernel<<<1024, 256, 8192, stream>>>(Yp, Y0, sbp, sb, wqc + i*3072,
                                                        Yw, st, (i > 0) ? 1 : 0);
    bn_finalize_kernel<<<1, 64, 0, stream>>>(st, bns_g + i*32, bns_b + i*32,
                                             sb + (1 + i)*128, 32, invNconv);
  }

  // ---- materialize final h = normq(Yw, sb9) + normq(Y0, sb0) ----
  norm2_kernel<<<8192, 256, 0, stream>>>(Yw, Y0, H, sb + 9*128, sb, BATCH*512/8);

  // ---- fc1 -> bn5 -> relu -> actq ----
  fc_wmma_kernel<<<256, 256, 0, stream>>>(H, wq1, Y1, stats + 10*128, 16, 512);
  bn_finalize_kernel<<<1, 64, 0, stream>>>(stats + 10*128, bn5_g, bn5_b, sb + 10*128, 64, invNfc);
  norm_actq_kernel<<<1024, 256, 0, stream>>>(Y1, A1, sb + 10*128, 64, BATCH*64/8);

  // ---- fc2 -> bn6 -> relu -> actq ----
  fc_wmma_kernel<<<256, 256, 0, stream>>>(A1, wq2, Y2, stats + 11*128, 2, 64);
  bn_finalize_kernel<<<1, 64, 0, stream>>>(stats + 11*128, bn6_g, bn6_b, sb + 11*128, 64, invNfc);
  norm_actq_kernel<<<1024, 256, 0, stream>>>(Y2, A2, sb + 11*128, 64, BATCH*64/8);

  // ---- fc3 + sigmoid ----
  fc3_sigmoid_kernel<<<BATCH/256, 256, 0, stream>>>(A2, qw3, fc3_b, out);
}
